// PointerNetwork_65317862637979
// MI455X (gfx1250) — matrix operand order repaired and checked by reading, not compiled, not run on hardware
//
#include <hip/hip_runtime.h>

// Problem constants (match reference)
#define Bn  32
#define Tn  1024
#define En  512
#define Hn  256
#define Sn  128
#define FourH 1024
#define TB  (Tn*Bn)   // 32768 GEMM rows

typedef _Float16 f16;
typedef unsigned int u32;
typedef __attribute__((ext_vector_type(16))) _Float16 v16h;
typedef __attribute__((ext_vector_type(8)))  float    v8f;
typedef __attribute__((ext_vector_type(4)))  u32      u32x4;

// ---------------------------------------------------------------------------
// WMMA fragment helpers (layouts per CDNA5 ISA 7.12.2, wave32)
//  A 16x32 f16: lane L -> row m = L&15 ; halves 0..7 = K[(L>>4)*8 .. +7],
//               halves 8..15 = K[16+(L>>4)*8 .. +7]     (two 16B loads)
//  B 32x16 f16 from row-major W[N][K]: lane L -> col n = L&15 ;
//               same K order as A                        (two 16B loads)
//  D 16x16 f32: lane L -> n = L&15 ; vgpr r -> m = (L>>4)*8 + r
// ---------------------------------------------------------------------------
union F16Frag { v16h v; u32x4 q[2]; };

__device__ __forceinline__ v16h load_a(const f16* __restrict__ row, int k, int hl) {
  F16Frag a;
  a.q[0] = *(const u32x4*)(row + k + hl * 8);
  a.q[1] = *(const u32x4*)(row + k + 16 + hl * 8);
  return a.v;
}
__device__ __forceinline__ v16h load_b(const f16* __restrict__ row, int k, int hl) {
  F16Frag b;
  b.q[0] = *(const u32x4*)(row + k + hl * 8);
  b.q[1] = *(const u32x4*)(row + k + 16 + hl * 8);
  return b.v;
}
__device__ __forceinline__ v8f wmma32(v16h a, v16h b, v8f c) {
  return __builtin_amdgcn_wmma_f32_16x16x32_f16(false, a, false, b, (short)0, c,
                                                false, false);
}
__device__ __forceinline__ float sigmoidf_(float x) {
  return 1.0f / (1.0f + __expf(-x));
}
// Defeat LICM: returns 0 as an opaque VGPR value the optimizer cannot fold.
// Adding it to a pointer inside a loop prevents the weight-fragment loads
// from being hoisted out of the 1024-step scan (which caused a 256-VGPR
// hoist -> scratch spill in an earlier build). "+v" on a 32-bit int is
// always legal for divergent values (the pointers themselves are per-lane).
__device__ __forceinline__ int opaque_zero_() {
  int z = 0;
  asm volatile("" : "+v"(z));
  return z;
}

// ---------------------------------------------------------------------------
// Generic WMMA GEMM:  C[M][N] = A[M][K] * W[N][K]^T (+ bias[n])
// block = 128 threads (4 waves); each wave computes 4 N-tiles sharing one A
// fragment (1 A load : 4 B loads : 4 WMMA). grid = (N/256, M/16)
// ---------------------------------------------------------------------------
template <typename OutT>
__global__ void gemm_wmma_k(const f16* __restrict__ A, const f16* __restrict__ W,
                            const float* __restrict__ bias, OutT* __restrict__ C,
                            int N, int K) {
  const int wave = threadIdx.x >> 5, lane = threadIdx.x & 31;
  const int hl = lane >> 4, ln = lane & 15;
  const int tn0 = (blockIdx.x * 4 + wave) * 4;  // first of 4 N-tiles
  const int tm = blockIdx.y;
  const f16* arow = A + (size_t)(tm * 16 + ln) * K;
  const f16* w0 = W + (size_t)((tn0 + 0) * 16 + ln) * K;
  const f16* w1 = W + (size_t)((tn0 + 1) * 16 + ln) * K;
  const f16* w2 = W + (size_t)((tn0 + 2) * 16 + ln) * K;
  const f16* w3 = W + (size_t)((tn0 + 3) * 16 + ln) * K;
  v8f a0 = {}, a1 = {}, a2 = {}, a3 = {};
  for (int k = 0; k < K; k += 32) {
    const v16h a = load_a(arow, k, hl);
    a0 = wmma32(a, load_b(w0, k, hl), a0);
    a1 = wmma32(a, load_b(w1, k, hl), a1);
    a2 = wmma32(a, load_b(w2, k, hl), a2);
    a3 = wmma32(a, load_b(w3, k, hl), a3);
  }
#pragma unroll
  for (int q = 0; q < 4; ++q) {
    const v8f acc = (q == 0) ? a0 : (q == 1) ? a1 : (q == 2) ? a2 : a3;
    const int n = (tn0 + q) * 16 + ln;
    const float bv = bias ? bias[n] : 0.0f;
#pragma unroll
    for (int r = 0; r < 8; ++r) {
      const int m = tm * 16 + hl * 8 + r;
      C[(size_t)m * N + n] = (OutT)(acc[r] + bv);
    }
  }
}

// ---------------------------------------------------------------------------
// Persistent encoder scan. One workgroup per direction, 1024 threads = 32
// waves. h lives in LDS (f16); per step: stage 64KB Xg block to LDS, do the
// recurrent GEMM h[32,256] x Whh^T[256,1024] with 4 WMMA accumulators per
// wave (i/f/g/o columns of the same j-tile), apply the LSTM cell in-register
// (c state stays in VGPRs across all 1024 steps). Weight fragments are
// re-loaded from global each step (L2-resident) via an opaque zero offset to
// prevent the compiler hoisting 256 VGPRs of fragments into scratch spills.
// ---------------------------------------------------------------------------
__global__ void enc_scan_k(const f16* __restrict__ Xg,   // [2][T*32][1024] rows m=t*32+b (bias folded in)
                           const f16* __restrict__ Whh,  // [2][1024][256]
                           f16* __restrict__ enc,        // [B][T][512] f16
                           f16* __restrict__ dh,         // [2][32][256] final h
                           float* __restrict__ dc) {     // [2][32][256] final c
  extern __shared__ char smem[];
  f16* xg_lds = (f16*)smem;                     // 32*1024 f16 = 64KB
  f16* h_lds  = (f16*)(smem + 32 * 1024 * 2);   // 32*256 f16  = 16KB
  const int d = blockIdx.x;
  const int tid = threadIdx.x;
  const int wave = tid >> 5, lane = tid & 31;
  const int hl = lane >> 4, ln = lane & 15;
  const int mt = wave >> 4;   // 0..1 : M tile (batch rows)
  const int jt = wave & 15;   // 0..15: hidden-column tile
  const f16* Wd = Whh + (size_t)d * 1024 * 256;
  const f16* Xd = Xg + (size_t)d * TB * 1024;
  const f16* w0 = Wd + (size_t)(0 * 256 + jt * 16 + ln) * 256;
  const f16* w1 = Wd + (size_t)(1 * 256 + jt * 16 + ln) * 256;
  const f16* w2 = Wd + (size_t)(2 * 256 + jt * 16 + ln) * 256;
  const f16* w3 = Wd + (size_t)(3 * 256 + jt * 16 + ln) * 256;
  const f16* arow = h_lds + (mt * 16 + ln) * 256;
  const int j = jt * 16 + ln;

  for (int i = tid; i < 32 * 256; i += 1024) h_lds[i] = (f16)0.0f;
  float creg[8];
#pragma unroll
  for (int r = 0; r < 8; ++r) creg[r] = 0.0f;

  for (int t = 0; t < Tn; ++t) {
    const int trow = d ? (Tn - 1 - t) : t;  // backward dir consumes reversed xs
    {  // stage this step's precomputed input gates (64KB) into LDS
      const u32x4* src = (const u32x4*)(Xd + (size_t)trow * 32 * 1024);
      u32x4* dst = (u32x4*)xg_lds;
      for (int i = tid; i < 4096; i += 1024) dst[i] = src[i];
      if (t + 1 < Tn) {  // software prefetch of next step's block (512B/thread)
        const int nrow = d ? (trow - 1) : (trow + 1);
        if (tid < 128)
          __builtin_prefetch(Xd + (size_t)nrow * 32 * 1024 + tid * 256, 0, 1);
      }
    }
    __syncthreads();

    // Opaque zero offset (fresh each iteration) keeps the 32 B-fragment
    // loads in-loop as global_load_b128 (L2 hits) instead of scratch reloads.
    const int z = opaque_zero_();
    const f16* w0l = w0 + z;
    const f16* w1l = w1 + z;
    const f16* w2l = w2 + z;
    const f16* w3l = w3 + z;

    v8f ai = {}, af = {}, ag = {}, ao = {};
    for (int k = 0; k < 256; k += 32) {
      const v16h a = load_a(arow, k, hl);
      ai = wmma32(a, load_b(w0l, k, hl), ai);
      af = wmma32(a, load_b(w1l, k, hl), af);
      ag = wmma32(a, load_b(w2l, k, hl), ag);
      ao = wmma32(a, load_b(w3l, k, hl), ao);
    }
    __syncthreads();  // all h_lds reads done before rewriting h

#pragma unroll
    for (int r = 0; r < 8; ++r) {
      const int bb = mt * 16 + hl * 8 + r;
      const float gi = ai[r] + (float)xg_lds[bb * 1024 + j];
      const float gf = af[r] + (float)xg_lds[bb * 1024 + 256 + j];
      const float gg = ag[r] + (float)xg_lds[bb * 1024 + 512 + j];
      const float go = ao[r] + (float)xg_lds[bb * 1024 + 768 + j];
      const float I = sigmoidf_(gi), F = sigmoidf_(gf);
      const float G = tanhf(gg), O = sigmoidf_(go);
      creg[r] = F * creg[r] + I * G;
      const float h = O * tanhf(creg[r]);
      h_lds[bb * 256 + j] = (f16)h;
      enc[((size_t)bb * Tn + trow) * 512 + (d << 8) + j] = (f16)h;
      if (t == Tn - 1) {
        dh[(d * 32 + bb) * 256 + j] = (f16)h;
        dc[(d * 32 + bb) * 256 + j] = creg[r];
      }
    }
    __syncthreads();  // xg_lds reads done before next staging
  }
}

// ---------------------------------------------------------------------------
// Decoder LSTM cell (one step, both directions). grid=2, block=1024.
// A = [dec_in | h_d] (32 x 768) staged in LDS; W = pre-concatenated
// [Wih|Whh] (1024 x 768). Same 4-accumulator gate fusion as the encoder.
// Note: hs_flat = h.reshape(B,-1) is the identity on the flat h buffer, so
// writing h16 contiguously *is* hs_flat[32][512].
// ---------------------------------------------------------------------------
__global__ void dec_cell_k(const f16* __restrict__ Wcat,  // [2][1024][768]
                           const float* __restrict__ bias,// [2][1024]
                           const f16* __restrict__ decin, // [32][512]
                           f16* __restrict__ h16,         // [2][32][256] in/out
                           float* __restrict__ c32) {     // [2][32][256] in/out
  extern __shared__ char smem[];
  f16* A = (f16*)smem;  // [32][768] = 48KB
  const int d = blockIdx.x;
  const int tid = threadIdx.x;
  // stage A = [decin | h_d], vectorized (8 halves per move)
  for (int i = tid; i < 2048; i += 1024) {  // 32*512/8 decin chunks
    const int m = i >> 6, k8 = (i & 63) * 8;
    *(u32x4*)(A + m * 768 + k8) = *(const u32x4*)(decin + m * 512 + k8);
  }
  for (int i = tid; i < 1024; i += 1024) {  // 32*256/8 h chunks
    const int m = i >> 5, k8 = (i & 31) * 8;
    *(u32x4*)(A + m * 768 + 512 + k8) =
        *(const u32x4*)(h16 + d * 8192 + m * 256 + k8);
  }
  __syncthreads();

  const int wave = tid >> 5, lane = tid & 31;
  const int hl = lane >> 4, ln = lane & 15;
  const int mt = wave >> 4, jt = wave & 15;
  const f16* Wd = Wcat + (size_t)d * 1024 * 768;
  const f16* w0 = Wd + (size_t)(0 * 256 + jt * 16 + ln) * 768;
  const f16* w1 = Wd + (size_t)(1 * 256 + jt * 16 + ln) * 768;
  const f16* w2 = Wd + (size_t)(2 * 256 + jt * 16 + ln) * 768;
  const f16* w3 = Wd + (size_t)(3 * 256 + jt * 16 + ln) * 768;
  const f16* arow = A + (mt * 16 + ln) * 768;
  v8f ai = {}, af = {}, ag = {}, ao = {};
  for (int k = 0; k < 768; k += 32) {
    const v16h a = load_a(arow, k, hl);
    ai = wmma32(a, load_b(w0, k, hl), ai);
    af = wmma32(a, load_b(w1, k, hl), af);
    ag = wmma32(a, load_b(w2, k, hl), ag);
    ao = wmma32(a, load_b(w3, k, hl), ao);
  }
  __syncthreads();  // staging reads of h16 complete before overwrite

  const int j = jt * 16 + ln;
  const float bi = bias[d * 1024 + j];
  const float bf = bias[d * 1024 + 256 + j];
  const float bg = bias[d * 1024 + 512 + j];
  const float bo = bias[d * 1024 + 768 + j];
#pragma unroll
  for (int r = 0; r < 8; ++r) {
    const int bb = mt * 16 + hl * 8 + r;
    const int cidx = d * 8192 + bb * 256 + j;
    const float I = sigmoidf_(ai[r] + bi);
    const float F = sigmoidf_(af[r] + bf);
    const float G = tanhf(ag[r] + bg);
    const float O = sigmoidf_(ao[r] + bo);
    float cc = F * c32[cidx] + I * G;
    c32[cidx] = cc;
    h16[cidx] = (f16)(O * tanhf(cc));  // contiguous => also hs_flat
  }
}

// ---------------------------------------------------------------------------
// Attention + softmax + argmax + mask update + gather. grid=B, block=256.
// scores[b,t] = v . tanh(dq[b]+eq[b,t]); mask==0 -> -1e9; softmax over t;
// peak = first argmax; mask[b,peak]=0; dec_in[b]=enc[b,peak].
// (sum(mask)==0 cannot occur: S=128 < T=1024 with unique removal.)
// ---------------------------------------------------------------------------
__global__ void attn_step_k(const float* __restrict__ dq,   // [32][256]
                            const float* __restrict__ eq,   // [B][T][256]
                            const float* __restrict__ av,   // [256]
                            float* __restrict__ mask,       // [32][1024] ws copy
                            const f16* __restrict__ enc,    // [B][T][512]
                            f16* __restrict__ decin,        // [32][512]
                            float* __restrict__ probs,      // [B][S][T]
                            float* __restrict__ peaks,      // [B][S]
                            int s) {
  __shared__ float dqs[256], vv[256], sc[1024], redv[256];
  __shared__ int redi[256];
  __shared__ int peak_s;
  const int b = blockIdx.x, tid = threadIdx.x;
  dqs[tid] = dq[b * 256 + tid];
  vv[tid] = av[tid];
  __syncthreads();

  for (int t = tid; t < Tn; t += 256) {
    const float* e = eq + ((size_t)b * Tn + t) * 256;
    float acc = 0.0f;
    for (int jj = 0; jj < 256; jj += 4) {
      const float4 ev = *(const float4*)(e + jj);
      acc += tanhf(dqs[jj] + ev.x) * vv[jj] +
             tanhf(dqs[jj + 1] + ev.y) * vv[jj + 1] +
             tanhf(dqs[jj + 2] + ev.z) * vv[jj + 2] +
             tanhf(dqs[jj + 3] + ev.w) * vv[jj + 3];
    }
    sc[t] = (mask[b * Tn + t] == 0.0f) ? -1e9f : acc;
  }
  __syncthreads();

  // max + first-index argmax
  float bv = -1e30f;
  int bi = 0;
  for (int t = tid; t < Tn; t += 256)
    if (sc[t] > bv) { bv = sc[t]; bi = t; }
  redv[tid] = bv; redi[tid] = bi;
  __syncthreads();
  for (int off = 128; off; off >>= 1) {
    if (tid < off) {
      const float ov = redv[tid + off]; const int oi = redi[tid + off];
      if (ov > redv[tid] || (ov == redv[tid] && oi < redi[tid])) {
        redv[tid] = ov; redi[tid] = oi;
      }
    }
    __syncthreads();
  }
  const float mx = redv[0];
  if (tid == 0) peak_s = redi[0];
  __syncthreads();

  // sum of exp
  float ls = 0.0f;
  for (int t = tid; t < Tn; t += 256) ls += __expf(sc[t] - mx);
  redv[tid] = ls;
  __syncthreads();
  for (int off = 128; off; off >>= 1) {
    if (tid < off) redv[tid] += redv[tid + off];
    __syncthreads();
  }
  const float inv = 1.0f / redv[0];
  for (int t = tid; t < Tn; t += 256)
    probs[((size_t)b * Sn + s) * Tn + t] = __expf(sc[t] - mx) * inv;

  const int pk = peak_s;
  if (tid == 0) {
    peaks[b * Sn + s] = (float)pk;
    mask[b * Tn + pk] = 0.0f;
  }
  const f16* er = enc + ((size_t)b * Tn + pk) * 512;
  decin[b * 512 + tid] = er[tid];
  decin[b * 512 + 256 + tid] = er[256 + tid];
}

// ------------------------- prep / conversion kernels -----------------------
__global__ void cvt_f16_k(const float* __restrict__ in, f16* __restrict__ out, int n) {
  const int i = blockIdx.x * 256 + threadIdx.x;
  if (i < n) out[i] = (f16)in[i];
}
__global__ void cvt_x_k(const float* __restrict__ x, f16* __restrict__ xs) {
  const int i = blockIdx.x * 256 + threadIdx.x;  // n = B*T*E = 16M
  if (i >= Bn * Tn * En) return;
  const int e = i & 511, r = i >> 9, bb = r >> 10, t = r & 1023;
  xs[(size_t)(t * 32 + bb) * 512 + e] = (f16)x[i];  // rows m = t*32 + b
}
__global__ void addb_k(const float* __restrict__ a, const float* __restrict__ b,
                       float* __restrict__ o, int n) {
  const int i = blockIdx.x * 256 + threadIdx.x;
  if (i < n) o[i] = a[i] + b[i];
}
__global__ void wcat_k(const float* __restrict__ wih, const float* __restrict__ whh,
                       f16* __restrict__ wc) {
  const int i = blockIdx.x * 256 + threadIdx.x;  // n = 2*1024*768
  if (i >= 2 * 1024 * 768) return;
  const int k = i % 768, n = (i / 768) & 1023, d = i / (768 * 1024);
  const float v = (k < 512) ? wih[((size_t)d * 1024 + n) * 512 + k]
                            : whh[((size_t)d * 1024 + n) * 256 + (k - 512)];
  wc[i] = (f16)v;
}
__global__ void initstate_k(const float* __restrict__ am, float* __restrict__ mws,
                            f16* __restrict__ din) {
  const int i = blockIdx.x * 256 + threadIdx.x;
  if (i < Bn * Tn) mws[i] = am[i];
  if (i < Bn * 512) din[i] = (f16)0.0f;
}

// ---------------------------------------------------------------------------
extern "C" void kernel_launch(void* const* d_in, const int* in_sizes, int n_in,
                              void* d_out, int out_size, void* d_ws, size_t ws_size,
                              hipStream_t stream) {
  const float* x       = (const float*)d_in[0];
  const float* amask   = (const float*)d_in[1];
  const float* enc_Wih = (const float*)d_in[2];
  const float* enc_Whh = (const float*)d_in[3];
  const float* enc_bih = (const float*)d_in[4];
  const float* enc_bhh = (const float*)d_in[5];
  const float* dec_Wih = (const float*)d_in[6];
  const float* dec_Whh = (const float*)d_in[7];
  const float* dec_bih = (const float*)d_in[8];
  const float* dec_bhh = (const float*)d_in[9];
  const float* att_Wd  = (const float*)d_in[10];
  const float* att_We  = (const float*)d_in[11];
  const float* att_v   = (const float*)d_in[12];
  (void)in_sizes; (void)n_in; (void)out_size; (void)ws_size;

  char* ws = (char*)d_ws;
  size_t o = 0;
  auto take = [&](size_t bytes) -> char* {
    char* p = ws + o;
    o = (o + bytes + 255) & ~(size_t)255;
    return p;
  };
  f16*   xs16    = (f16*)take((size_t)TB * En * 2);          // 33.5 MB
  f16*   wih16   = (f16*)take(2ull * 1024 * 512 * 2);
  f16*   whh16   = (f16*)take(2ull * 1024 * 256 * 2);
  f16*   wcat16  = (f16*)take(2ull * 1024 * 768 * 2);
  f16*   we16    = (f16*)take(256ull * 512 * 2);
  f16*   wd16    = (f16*)take(256ull * 512 * 2);
  float* encb    = (float*)take(2048 * 4);
  float* decb    = (float*)take(2048 * 4);
  f16*   xg16    = (f16*)take(2ull * TB * 1024 * 2);         // 134 MB
  f16*   enc16   = (f16*)take((size_t)Bn * Tn * 512 * 2);    // 33.5 MB
  float* eq      = (float*)take((size_t)Bn * Tn * 256 * 4);  // 33.5 MB
  f16*   dh16    = (f16*)take(2ull * 32 * 256 * 2);          // == hs_flat
  float* dc      = (float*)take(2ull * 32 * 256 * 4);
  f16*   decin16 = (f16*)take(32ull * 512 * 2);
  float* dq      = (float*)take(32ull * 256 * 4);
  float* maskws  = (float*)take(32ull * 1024 * 4);

  float* probs = (float*)d_out;
  float* peaks = probs + (size_t)Bn * Sn * Tn;

  auto nb = [](int n) { return dim3((unsigned)((n + 255) / 256)); };

  // --- prep: f16 conversions, bias sums, concat decoder weights, state init
  cvt_f16_k<<<nb(2 * 1024 * 512), 256, 0, stream>>>(enc_Wih, wih16, 2 * 1024 * 512);
  cvt_f16_k<<<nb(2 * 1024 * 256), 256, 0, stream>>>(enc_Whh, whh16, 2 * 1024 * 256);
  cvt_f16_k<<<nb(256 * 512), 256, 0, stream>>>(att_We, we16, 256 * 512);
  cvt_f16_k<<<nb(256 * 512), 256, 0, stream>>>(att_Wd, wd16, 256 * 512);
  cvt_x_k<<<nb(Bn * Tn * En), 256, 0, stream>>>(x, xs16);
  addb_k<<<nb(2048), 256, 0, stream>>>(enc_bih, enc_bhh, encb, 2048);
  addb_k<<<nb(2048), 256, 0, stream>>>(dec_bih, dec_bhh, decb, 2048);
  wcat_k<<<nb(2 * 1024 * 768), 256, 0, stream>>>(dec_Wih, dec_Whh, wcat16);
  initstate_k<<<nb(Bn * Tn), 256, 0, stream>>>(amask, maskws, decin16);

  // --- phase 1: big parallel GEMM  Xg[d] = xs @ Wih[d]^T + bias  (f16 out)
  for (int d = 0; d < 2; ++d)
    gemm_wmma_k<f16><<<dim3(FourH / 256, TB / 16), 128, 0, stream>>>(
        xs16, wih16 + (size_t)d * 1024 * 512, encb + d * 1024,
        xg16 + (size_t)d * TB * 1024, FourH, En);

  // --- phase 2: persistent bidirectional encoder scan (WMMA recurrent GEMM)
  enc_scan_k<<<2, 1024, 32 * 1024 * 2 + 32 * 256 * 2, stream>>>(xg16, whh16,
                                                                enc16, dh16, dc);

  // --- phase 3: eq = enc @ We^T  (step-independent attention keys)
  gemm_wmma_k<float><<<dim3(256 / 256, TB / 16), 128, 0, stream>>>(
      enc16, we16, nullptr, eq, 256, 512);

  // --- phase 4: decoder loop
  for (int s = 0; s < Sn; ++s) {
    dec_cell_k<<<2, 1024, 32 * 768 * 2, stream>>>(wcat16, decb, decin16, dh16, dc);
    gemm_wmma_k<float><<<dim3(1, 2), 128, 0, stream>>>(
        dh16 /* == hs_flat[32][512] */, wd16, nullptr, dq, 256, 512);
    attn_step_k<<<Bn, 256, 0, stream>>>(dq, eq, att_v, maskws, enc16, decin16,
                                        probs, peaks, s);
  }
}
